// SelfAttention_21053929685124
// MI455X (gfx1250) — matrix operand order repaired and checked by reading, not compiled, not run on hardware
//
#include <hip/hip_runtime.h>

// Problem constants (match reference)
#define BB   4
#define SS   2048
#define EE   768
#define DD   768
#define MRAc 64
#define NEGV 1.0e9f
// Band window: keys in [q-64, q+64]; per 16-query tile the union window is
// [q0-64, q0+79] = 144 keys = 9 key tiles of 16. Pad to 160 for K=32 chunks.
#define WIN  144
#define WINP 160

typedef __attribute__((ext_vector_type(16))) __bf16 v16bf;
typedef __attribute__((ext_vector_type(8)))  __bf16 v8bf;
typedef __attribute__((ext_vector_type(8)))  float  v8f;

// ---------------------------------------------------------------------------
// Async global->LDS staging (CDNA5 GLOBAL_LOAD_ASYNC_TO_LDS_B128, ASYNCcnt).
// ISA 15.18: the LDS address includes the instruction offset and the global
// address also adds INST_OFFSET -> one base address pair serves a 64B run
// (4 chunks) when LDS and global are both contiguous, via offset:0/16/32/48.
// Flip USE_ASYNC to 0 to fall back to load+store staging.
// ---------------------------------------------------------------------------
#define USE_ASYNC 1

#if USE_ASYNC
typedef __attribute__((address_space(3))) void       lds_t;
typedef __attribute__((address_space(1))) const void gmem_t;

__device__ __forceinline__ unsigned lds_a(const void* p) {
    return (unsigned)(unsigned long long)(lds_t*)p;
}
__device__ __forceinline__ unsigned long long gbl_a(const void* p) {
    return (unsigned long long)(gmem_t*)p;
}
#define G2L_B128_OFF(L, G, OFF)                                            \
    asm volatile("global_load_async_to_lds_b128 %0, %1, off offset:" #OFF  \
                 :: "v"(L), "v"(G) : "memory")

// one 16B chunk
__device__ __forceinline__ void g2l_b128(void* lds_dst, const void* gsrc) {
    unsigned L = lds_a(lds_dst);
    unsigned long long G = gbl_a(gsrc);
    G2L_B128_OFF(L, G, 0);
}
// 64 contiguous bytes (4 chunks, shared base address)
__device__ __forceinline__ void g2l_64B(void* lds_dst, const void* gsrc) {
    unsigned L = lds_a(lds_dst);
    unsigned long long G = gbl_a(gsrc);
    G2L_B128_OFF(L, G, 0);
    G2L_B128_OFF(L, G, 16);
    G2L_B128_OFF(L, G, 32);
    G2L_B128_OFF(L, G, 48);
}
__device__ __forceinline__ void async_join()  {   // all copies landed
    asm volatile("s_wait_asynccnt 0x0" ::: "memory");
}
__device__ __forceinline__ void async_join8() {   // oldest copies landed, 8 may fly
    asm volatile("s_wait_asynccnt 0x8" ::: "memory");
}
#else
__device__ __forceinline__ void g2l_b128(void* lds_dst, const void* gsrc) {
    *(uint4*)lds_dst = *(const uint4*)gsrc;
}
__device__ __forceinline__ void g2l_64B(void* lds_dst, const void* gsrc) {
#pragma unroll
    for (int q = 0; q < 4; ++q)
        ((uint4*)lds_dst)[q] = ((const uint4*)gsrc)[q];
}
__device__ __forceinline__ void async_join()  {}
__device__ __forceinline__ void async_join8() {}
#endif

// ---------------------------------------------------------------------------
// Fragment loaders.  All LDS tiles are K-major, so every fragment is two
// contiguous 16B LDS loads (ds_load_b128) + a register shuffle.
// A (16x32): elems 0..7 at [base+half*8], elems 8..15 at [base+16+half*8].
// B (32x16): elems 0..7 at [base+half*8], elems 8..15 at [base+16+half*8].
// ---------------------------------------------------------------------------
__device__ __forceinline__ v16bf load_a_frag(const __bf16* p, int half) {
    v8bf lo = *(const v8bf*)(p + (half << 3));
    v8bf hi = *(const v8bf*)(p + 16 + (half << 3));
    return __builtin_shufflevector(lo, hi, 0, 1, 2, 3, 4, 5, 6, 7,
                                           8, 9, 10, 11, 12, 13, 14, 15);
}
__device__ __forceinline__ v16bf load_b_frag(const __bf16* p, int half) {
    v8bf lo = *(const v8bf*)(p + (half << 3));
    v8bf hi = *(const v8bf*)(p + 16 + (half << 3));
    return __builtin_shufflevector(lo, hi, 0, 1, 2, 3, 4, 5, 6, 7,
                                           8, 9, 10, 11, 12, 13, 14, 15);
}

// ---------------------------------------------------------------------------
// Kernel A1: fp32 -> bf16 conversion (x), grid-stride
// ---------------------------------------------------------------------------
__global__ __launch_bounds__(256) void cvt_bf16_kernel(const float* __restrict__ src,
                                                       __bf16* __restrict__ dst, int n) {
    int i = blockIdx.x * blockDim.x + threadIdx.x;
    int stride = gridDim.x * blockDim.x;
    for (; i < n; i += stride) dst[i] = (__bf16)src[i];
}

// ---------------------------------------------------------------------------
// Kernel A2: fp32 -> bf16 + transpose for the three weight matrices.
// Wt[n][k] = W[k][n].  Tiled through LDS so both global sides are coalesced.
// grid = (24, 24, 3), block = 256 (32x8 logical)
// ---------------------------------------------------------------------------
__global__ __launch_bounds__(256) void cvt_tr_kernel(
    const float* __restrict__ Wq, const float* __restrict__ Wk,
    const float* __restrict__ Wv,
    __bf16* __restrict__ Wqt, __bf16* __restrict__ Wkt, __bf16* __restrict__ Wvt) {

    const float* src = (blockIdx.z == 0) ? Wq : (blockIdx.z == 1) ? Wk : Wv;
    __bf16*      dst = (blockIdx.z == 0) ? Wqt : (blockIdx.z == 1) ? Wkt : Wvt;

    __shared__ __bf16 tile[32][33];   // +1 pad: avoid bank conflicts
    const int k0 = blockIdx.x * 32;
    const int n0 = blockIdx.y * 32;
    const int tx = threadIdx.x & 31;
    const int ty = threadIdx.x >> 5;  // 0..7

#pragma unroll
    for (int j = 0; j < 4; ++j)
        tile[ty + 8 * j][tx] = (__bf16)src[(size_t)(k0 + ty + 8 * j) * DD + n0 + tx];
    __syncthreads();
#pragma unroll
    for (int j = 0; j < 4; ++j)
        dst[(size_t)(n0 + ty + 8 * j) * DD + k0 + tx] = tile[tx][ty + 8 * j];
}

// ---------------------------------------------------------------------------
// Kernel B: generic NT GEMM, C[M,N] = A[M,768] @ Bt[N,768]^T  (bf16 in/out,
// f32 accumulate).  Double-buffered async pipeline: issue tile k+1, partial
// wait (asynccnt<=8) so tile k+1 copies overlap the WMMAs on tile k.
// grid = (M/64, N/64), block = 128 = 4 waves.  Wave w owns N-subtile w.
// Used for:  Q = X@Wqt^T,  K = X@Wkt^T,  Vt = Wvt@X^T.
// ---------------------------------------------------------------------------
__global__ __launch_bounds__(128) void gemm_nt_kernel(
    const __bf16* __restrict__ A, const __bf16* __restrict__ Bt,
    __bf16* __restrict__ C, int ldc) {

    const int m0 = blockIdx.x * 64;
    const int n0 = blockIdx.y * 64;

    __shared__ __align__(16) __bf16 sA[2][64 * 64];   // 2 x 8 KB
    __shared__ __align__(16) __bf16 sB[2][64 * 64];   // 2 x 8 KB

    const int tid  = threadIdx.x;
    const int wv   = tid >> 5;      // wave 0..3 (wave32)
    const int lane = tid & 31;
    const int half = lane >> 4;
    const int lrow = lane & 15;

    // each thread owns one 64B run per tile: row r, columns c..c+31
    const int cr = tid >> 1;             // 0..63
    const int cc = (tid & 1) << 5;       // 0 or 32

    // 4 + 4 async chunks per thread per tile (matches asynccnt<=8 wait)
    auto issue = [&](int t, int buf) {
        int k0 = t * 64;
        g2l_64B(&sA[buf][cr * 64 + cc], &A[(size_t)(m0 + cr) * DD + k0 + cc]);
        g2l_64B(&sB[buf][cr * 64 + cc], &Bt[(size_t)(n0 + cr) * DD + k0 + cc]);
    };

    v8f acc[4] = {};
    issue(0, 0);

    for (int kc = 0; kc < 12; ++kc) {
        __syncthreads();                   // everyone done reading buf[(kc+1)&1]
        if (kc + 1 < 12) {
            issue(kc + 1, (kc + 1) & 1);
            __builtin_prefetch(&A[(size_t)(m0 + cr) * DD + kc * 64 + 128], 0, 3);
            async_join8();                 // tile kc landed; tile kc+1 in flight
        } else {
            async_join();
        }
        __syncthreads();                   // tile kc visible to all waves

        const __bf16* a = sA[kc & 1];
        const __bf16* b = sB[kc & 1];
#pragma unroll
        for (int s = 0; s < 2; ++s) {
            v16bf bfr = load_b_frag(&b[((wv << 4) + lrow) * 64 + s * 32], half);
#pragma unroll
            for (int mt = 0; mt < 4; ++mt) {
                v16bf afr = load_a_frag(&a[(mt * 16 + lrow) * 64 + s * 32], half);
                acc[mt] = __builtin_amdgcn_wmma_f32_16x16x32_bf16(
                    false, afr, false, bfr, (short)0, acc[mt], false, false);
            }
        }
    }

    // epilogue: C layout m = r + half*8, n = lane&15
#pragma unroll
    for (int mt = 0; mt < 4; ++mt)
#pragma unroll
        for (int r = 0; r < 8; ++r) {
            int gm = m0 + mt * 16 + r + half * 8;
            int gn = n0 + (wv << 4) + lrow;
            C[(size_t)gm * ldc + gn] = (__bf16)acc[mt][r];
        }
}

// ---------------------------------------------------------------------------
// Kernel C: banded attention.  One block per (batch, 16-query tile).
// block = 256 = 8 waves.  Wave w owns score key-tile w (wave 0 also tile 8).
// V is consumed transposed (Vt[e][token]) so the P@V B-fragments are K-major.
// LDS (59 KB): [sP 5K][ sS 10K | sQ 24K | sK 20K   <-- overlaid by sVt 48K ]
// ---------------------------------------------------------------------------
__global__ __launch_bounds__(256) void banded_attn_kernel(
    const __bf16* __restrict__ Qb, const __bf16* __restrict__ Kb,
    const __bf16* __restrict__ Vt, const int* __restrict__ unk,
    const float* __restrict__ rel, float* __restrict__ Out) {

    const int blk    = blockIdx.x;
    const int bidx   = blk >> 7;          // / 128 query tiles
    const int q0     = (blk & 127) * 16;
    const int kbase0 = q0 - MRAc;         // first key of the 144-wide window

    __shared__ __align__(16) unsigned char smem[5120 + 55296];
    __bf16* sP  = (__bf16*)smem;                        // 16 x 160 bf16 (5 KB)
    float*  sS  = (float*)(smem + 5120);                // 16 x 160 f32 (10 KB)
    __bf16* sQ  = (__bf16*)(smem + 15360);              // 16 x 768 bf16 (24 KB)
    __bf16* sK  = (__bf16*)(smem + 39936);              // 160 x 64 bf16 (20 KB)
    __bf16* sVt = (__bf16*)(smem + 5120);               // 768 x 32 bf16 (48 KB, stage-4 overlay)

    const int tid  = threadIdx.x;
    const int wv   = tid >> 5;      // wave 0..7
    const int lane = tid & 31;
    const int half = lane >> 4;
    const int lrow = lane & 15;

    // ---- stage 1: Q tile -> LDS (async; joined inside stage-2 iteration 0) --
    {
        const __bf16* qsrc = Qb + ((size_t)(bidx * SS + q0)) * DD;
        // 16 rows x 24 64B-runs = 384 groups
#pragma unroll
        for (int j = 0; j < 2; ++j) {
            int g = tid + 256 * j;
            if (g < 384) {
                int r = g / 24, c = (g % 24) << 5;
                g2l_64B(&sQ[r * DD + c], &qsrc[(size_t)r * DD + c]);
            }
        }
    }

    // ---- stage 2: raw scores S[16,144] = Qtile @ Kwin^T --------------------
    v8f acc0 = {}, acc1 = {};
    for (int kc = 0; kc < 12; ++kc) {
        __syncthreads();
        // sK: 160 rows x 2 64B-runs = 320 groups; row-level validity
#pragma unroll
        for (int j = 0; j < 2; ++j) {
            int g = tid + 256 * j;
            if (g < 320) {
                int r = g >> 1, c = (g & 1) << 5;
                int key = kbase0 + r;
                if (r < WIN && key >= 0 && key < SS) {
                    g2l_64B(&sK[r * 64 + c],
                            &Kb[((size_t)(bidx * SS + key)) * DD + kc * 64 + c]);
                } else {
                    uint4 z{};
#pragma unroll
                    for (int q = 0; q < 4; ++q)
                        ((uint4*)&sK[r * 64 + c])[q] = z;
                }
            }
        }
        async_join();                                // also joins stage-1 sQ copies
        __syncthreads();

#pragma unroll
        for (int s = 0; s < 2; ++s) {
            v16bf afr = load_a_frag(&sQ[lrow * DD + kc * 64 + s * 32], half);
            v16bf bfr = load_b_frag(&sK[(wv * 16 + lrow) * 64 + s * 32], half);
            acc0 = __builtin_amdgcn_wmma_f32_16x16x32_bf16(
                false, afr, false, bfr, (short)0, acc0, false, false);
            if (wv == 0) {                           // wave-uniform: EXEC stays full
                v16bf bfr2 = load_b_frag(&sK[(128 + lrow) * 64 + s * 32], half);
                acc1 = __builtin_amdgcn_wmma_f32_16x16x32_bf16(
                    false, afr, false, bfr2, (short)0, acc1, false, false);
            }
        }
    }
#pragma unroll
    for (int r = 0; r < 8; ++r)
        sS[(r + half * 8) * WINP + wv * 16 + lrow] = acc0[r];
    if (wv == 0) {
#pragma unroll
        for (int r = 0; r < 8; ++r)
            sS[(r + half * 8) * WINP + 128 + lrow] = acc1[r];
    }
    __syncthreads();

    // ---- stage 3: mask + relative bias + softmax (fp32) --------------------
    {
        const int row = tid >> 4;           // 16 threads per query row
        const int sub = tid & 15;
        const float scale = 0.03608439182435161f;    // 1/sqrt(768)
        float mx = -3.0e38f;
        for (int c = sub; c < WIN; c += 16) {
            int key = kbase0 + c;
            float s = -3.0e38f;
            if (key >= 0 && key < SS) {
                float raw = sS[row * WINP + c] * scale;
                int d = c - 64 - row;                // key - query
                float pen = 0.0f;
                if (d < -MRAc || d > MRAc) pen += NEGV;          // band mask
                if (unk[bidx * SS + key] == 0) pen += NEGV;      // key padding mask
                int ridx = c - row;                  // clip(j - i + 64, 0, 128)
                ridx = ridx < 0 ? 0 : (ridx > 2 * MRAc ? 2 * MRAc : ridx);
                s = raw - pen + rel[ridx];
            }
            sS[row * WINP + c] = s;
            mx = fmaxf(mx, s);
        }
#pragma unroll
        for (int off = 8; off >= 1; off >>= 1)       // 16-lane groups of a wave32
            mx = fmaxf(mx, __shfl_xor(mx, off, 16));
        float sum = 0.0f;
        for (int c = sub; c < WIN; c += 16)
            sum += __expf(sS[row * WINP + c] - mx);
#pragma unroll
        for (int off = 8; off >= 1; off >>= 1)
            sum += __shfl_xor(sum, off, 16);
        float inv = 1.0f / sum;                      // diagonal key always valid
        for (int c = sub; c < WINP; c += 16) {
            float p = (c < WIN) ? __expf(sS[row * WINP + c] - mx) * inv : 0.0f;
            sP[row * WINP + c] = (__bf16)p;          // zero padding cols 144..159
        }
    }

    // ---- stage 4: Out[16,768] = P[16,160] @ Vwin[160,768] ------------------
    // sVt tile is [e][kk]: K-major B fragments.  One 64B run per e-row; the
    // validity boundaries (0, 144, 2048) are multiples of 8 keys, so each 16B
    // chunk is entirely valid or invalid (chunk-level guards).
    v8f accO[6] = {};
    for (int kc = 0; kc < 5; ++kc) {
        __syncthreads();   // also fences softmax's sS reads before sVt overlay
        const int kk0 = kc * 32;
#pragma unroll
        for (int j = 0; j < 3; ++j) {
            int e = tid + 256 * j;                   // 768 rows, 1 run each
            __bf16* ldst = &sVt[e * 32];
            const __bf16* gsrc =
                &Vt[(size_t)e * (BB * SS) + (size_t)bidx * SS + kbase0 + kk0];
#if USE_ASYNC
            unsigned L = lds_a(ldst);
            unsigned long long G = gbl_a(gsrc);
#endif
            uint4 z{};
#pragma unroll
            for (int q = 0; q < 4; ++q) {
                int kk  = kk0 + (q << 3);
                int key = kbase0 + kk;
                bool ok = (kk < WIN) && (key >= 0) && (key < SS);
#if USE_ASYNC
                if (ok) {
                    switch (q) {                     // literal offsets
                        case 0: G2L_B128_OFF(L, G, 0);  break;
                        case 1: G2L_B128_OFF(L, G, 16); break;
                        case 2: G2L_B128_OFF(L, G, 32); break;
                        default: G2L_B128_OFF(L, G, 48); break;
                    }
                } else {
                    ((uint4*)ldst)[q] = z;
                }
#else
                if (ok) ((uint4*)ldst)[q] = ((const uint4*)gsrc)[q];
                else    ((uint4*)ldst)[q] = z;
#endif
            }
        }
        async_join();
        __syncthreads();

        v16bf afr = load_a_frag(&sP[lrow * WINP + kc * 32], half);
#pragma unroll
        for (int j = 0; j < 6; ++j) {
            int nt = wv + 8 * j;                     // 48 N-tiles / 8 waves
            v16bf bfr = load_b_frag(&sVt[(nt * 16 + lrow) * 32], half);
            accO[j] = __builtin_amdgcn_wmma_f32_16x16x32_bf16(
                false, afr, false, bfr, (short)0, accO[j], false, false);
        }
    }

#pragma unroll
    for (int j = 0; j < 6; ++j) {
        int nt = wv + 8 * j;
#pragma unroll
        for (int r = 0; r < 8; ++r)
            Out[((size_t)(bidx * SS + q0 + r + half * 8)) * EE + nt * 16 + lrow] =
                accO[j][r];
    }
}

// ---------------------------------------------------------------------------
// Host-side launcher
// inputs: x[B,S,E] f32, unk_mask[B,S] i32, Wq[E,D] f32, Wk[E,D] f32,
//         Wv[E,E] f32, rel[129] f32;  output: [B,S,E] f32
// workspace: Xb | Wqt | Wkt | Wvt | Qb | Kb | Vt  (all bf16, ~54 MB)
// ---------------------------------------------------------------------------
extern "C" void kernel_launch(void* const* d_in, const int* in_sizes, int n_in,
                              void* d_out, int out_size, void* d_ws, size_t ws_size,
                              hipStream_t stream) {
    const float* x   = (const float*)d_in[0];
    const int*   unk = (const int*)d_in[1];
    const float* Wq  = (const float*)d_in[2];
    const float* Wk  = (const float*)d_in[3];
    const float* Wv  = (const float*)d_in[4];
    const float* rel = (const float*)d_in[5];
    float*       out = (float*)d_out;

    char* ws = (char*)d_ws;
    const size_t NX = (size_t)BB * SS * EE;   // 6,291,456 tokens x E
    const size_t NW = (size_t)EE * DD;        //   589,824
    __bf16* Xb  = (__bf16*)(ws);                          // [8192][768]
    __bf16* Wqt = (__bf16*)(ws + NX * 2);                 // [768 n][768 k]
    __bf16* Wkt = (__bf16*)(ws + NX * 2 + NW * 2);
    __bf16* Wvt = (__bf16*)(ws + NX * 2 + NW * 4);
    __bf16* Qb  = (__bf16*)(ws + NX * 2 + NW * 6);        // [8192][768]
    __bf16* Kb  = (__bf16*)(ws + NX * 4 + NW * 6);        // [8192][768]
    __bf16* Vtb = (__bf16*)(ws + NX * 6 + NW * 6);        // [768 e][8192 tok]

    cvt_bf16_kernel<<<2048, 256, 0, stream>>>(x, Xb, (int)NX);
    cvt_tr_kernel<<<dim3(24, 24, 3), 256, 0, stream>>>(Wq, Wk, Wv, Wqt, Wkt, Wvt);

    // Q = X @ Wq,  K = X @ Wk  (both as A[M,K] x Bt[N,K]^T)
    gemm_nt_kernel<<<dim3(128, 12), 128, 0, stream>>>(Xb, Wqt, Qb, DD);
    gemm_nt_kernel<<<dim3(128, 12), 128, 0, stream>>>(Xb, Wkt, Kb, DD);
    // Vt[e][token] = (X @ Wv)^T = Wvt[e,:k] . X[token,:k]^T
    gemm_nt_kernel<<<dim3(12, 128), 128, 0, stream>>>(Wvt, Xb, Vtb, BB * SS);

    banded_attn_kernel<<<BB * (SS / 16), 256, 0, stream>>>(Qb, Kb, Vtb, unk, rel, out);
}
